// MambaMoE_68659347194406
// MI455X (gfx1250) — matrix operand order repaired and checked by reading, not compiled, not run on hardware
//
#include <hip/hip_runtime.h>
#include <hip/hip_bf16.h>
#include <math.h>

typedef _Float16 f16;
typedef __attribute__((ext_vector_type(16))) _Float16 v16h;
typedef __attribute__((ext_vector_type(8)))  float    v8f;

#define BB   32
#define CC   64
#define HH   64
#define WW_  64
#define HWW  4096
#define EE   4
#define DD   (CC*HWW)      // 262144

// output patch per workgroup: 8 rows x 16 cols
#define PH 8
#define PW 16

// conv tile geometry (two stacked 3x3 convs)
#define XR 12
#define XC 20
#define XN (XR*XC)         // 240 x-positions (halo 2)
#define HR 10
#define HC 18
#define HN (HR*HC)         // 180 h-positions (halo 1)

#define S2 72              // padded LDS row stride (halfs): 144B -> bank-conflict-free

// workspace layout (bytes)
#define WS_TG   0                          // float[2][32][2]
#define WS_TI   512                        // int[2][32][2]
#define WS_PW   4096                       // f16[8][2][9][64][64]
#define PW_HALFS (8*2*9*64*64)             // 589824
#define WS_PWR  (WS_PW + PW_HALFS*2)       // 1183744, f16[9][64][128]
#define PWR_HALFS (9*64*128)               // 73728
#define WS_MID  (2*1024*1024)              // f16[2][32][64][64][64]

// gfx1250 async global->LDS path (guarded; falls back to reg-staged copy)
#if defined(__AMDGCN__) && __has_builtin(__builtin_amdgcn_global_load_async_to_lds_b128) && __has_builtin(__builtin_amdgcn_s_wait_asynccnt)
#define HAVE_ASYNC_LDS 1
typedef __attribute__((ext_vector_type(4))) int i4v;
typedef __attribute__((address_space(1))) i4v g_i4v;   // global int4*
typedef __attribute__((address_space(3))) i4v l_i4v;   // LDS int4*
#else
#define HAVE_ASYNC_LDS 0
#endif

// ---------------- WMMA fragment loaders (CDNA5 16x16x32 f16 layouts) ----------
// A (MxK = 16x32): lanes 0-15 hold M=lane, K {0..7,16..23}; lanes 16-31 hold
// M=lane-16, K {8..15,24..31}.  row_base points at K-contiguous row of M.
__device__ __forceinline__ v16h load_fragA(const f16* row_base, int kc, int lane) {
  int klo = (lane & 16) ? 8 : 0;
  union { v16h v; uint4 q[2]; } u;
  u.q[0] = *(const uint4*)(row_base + kc + klo);
  u.q[1] = *(const uint4*)(row_base + kc + 16 + klo);
  return u.v;
}
// B (KxN = 32x16): lanes 0-15 hold N=lane, K {0..7,16..23}; lanes 16-31 hold N=lane-16,
// K {8..15,24..31}.  col_base points at K-contiguous column of N.
__device__ __forceinline__ v16h load_fragB(const f16* col_base, int kc, int lane) {
  int klo = (lane & 16) ? 8 : 0;
  union { v16h v; uint4 q[2]; } u;
  u.q[0] = *(const uint4*)(col_base + kc + klo);
  u.q[1] = *(const uint4*)(col_base + kc + 16 + klo);
  return u.v;
}

// ---------------- Kernel 1: noisy top-2 gating -------------------------------
__global__ __launch_bounds__(256) void moe_gate_kernel(
    const float* __restrict__ xr, const float* __restrict__ xi,
    const float* __restrict__ nr, const float* __restrict__ ni,
    const float* __restrict__ wgr, const float* __restrict__ wnr,
    const float* __restrict__ wgi, const float* __restrict__ wni,
    float* __restrict__ tg, int* __restrict__ ti)
{
  int br = blockIdx.x >> 5;
  int b  = blockIdx.x & 31;
  const float* x  = br ? xi  : xr;
  const float* wg = br ? wgi : wgr;
  const float* wn = br ? wni : wnr;
  const float* nz = br ? ni  : nr;
  int tid = threadIdx.x;

  float cg[4] = {0.f,0.f,0.f,0.f}, cn[4] = {0.f,0.f,0.f,0.f};
  const float* xb = x + (size_t)b * DD;
  for (int i = tid; i < DD; i += 256) {
    float xv = xb[i];
    const float* g4 = wg + (size_t)i * 4;
    const float* n4 = wn + (size_t)i * 4;
    #pragma unroll
    for (int e = 0; e < 4; e++) {
      cg[e] = fmaf(xv, g4[e], cg[e]);
      cn[e] = fmaf(xv, n4[e], cn[e]);
    }
  }
  __shared__ float red[8][256];
  #pragma unroll
  for (int e = 0; e < 4; e++) { red[e][tid] = cg[e]; red[4+e][tid] = cn[e]; }
  __syncthreads();
  for (int s = 128; s > 0; s >>= 1) {
    if (tid < s) {
      #pragma unroll
      for (int e = 0; e < 8; e++) red[e][tid] += red[e][tid + s];
    }
    __syncthreads();
  }
  if (tid == 0) {
    float logit[4];
    #pragma unroll
    for (int e = 0; e < 4; e++) {
      float z  = red[4+e][0];
      float sp = (z > 20.f) ? z : log1pf(expf(z));
      logit[e] = red[e][0] + nz[b*4 + e] * (sp + 1e-2f);
    }
    int i1 = 0;
    for (int e = 1; e < 4; e++) if (logit[e] > logit[i1]) i1 = e;
    int i2 = (i1 == 0) ? 1 : 0;
    for (int e = 0; e < 4; e++) if (e != i1 && logit[e] > logit[i2]) i2 = e;
    float m  = fmaxf(logit[i1], logit[i2]);
    float e1 = expf(logit[i1] - m), e2 = expf(logit[i2] - m);
    float s  = e1 + e2;
    int o = (br*BB + b) * 2;
    tg[o] = e1/s; tg[o+1] = e2/s;
    ti[o] = i1;   ti[o+1] = i2;
  }
}

// ---------------- Kernel 2: weight repack (f32 -> f16, tap-major) ------------
__global__ __launch_bounds__(256) void moe_pack_kernel(
    const float* __restrict__ We1, const float* __restrict__ We2,
    const float* __restrict__ Wr,  f16* __restrict__ pw, f16* __restrict__ pwr)
{
  int n = blockIdx.x * 256 + threadIdx.x;
  if (n < PW_HALFS) {
    // pw[e][layer][tap][o][i]  <-  We[layer][e][o][i][kh][kw]
    int e   = n / 73728;
    int r1  = n % 73728;
    int lyr = r1 / 36864;
    int r2  = r1 % 36864;
    int tap = r2 / 4096;
    int o   = (r2 >> 6) & 63;
    int i   = r2 & 63;
    const float* src = lyr ? We2 : We1;
    pw[n] = (f16)src[(((size_t)(e*64 + o))*64 + i)*9 + tap];
  } else if (n < PW_HALFS + PWR_HALFS) {
    int m   = n - PW_HALFS;       // pwr[tap][o][i128] <- Wr[o][i][kh][kw]
    int tap = m / 8192;
    int o   = (m >> 7) & 63;
    int i   = m & 127;
    pwr[m] = (f16)Wr[((size_t)(o*128 + i))*9 + tap];
  }
}

// ---------------- Kernel 3: MoE expert convs (top-2 only), WMMA --------------
__global__ __launch_bounds__(256) void moe_expert_kernel(
    const float* __restrict__ xr, const float* __restrict__ xi,
    const float* __restrict__ be1, const float* __restrict__ be2,
    const f16* __restrict__ pw,
    const float* __restrict__ tg, const int* __restrict__ ti,
    f16* __restrict__ mid)
{
  int blk = blockIdx.x;
  int br  = blk >> 10;
  int rem = blk & 1023;
  int b   = rem >> 5;
  int pi  = rem & 31;
  int oh0 = (pi >> 2) * PH;
  int ow0 = (pi & 3) * PW;
  const float* x = br ? xi : xr;

  int tid  = threadIdx.x;
  int wave = tid >> 5;
  int lane = tid & 31;
  int l    = lane & 15;
  int hi8  = (lane >> 4) << 3;     // 0 or 8: which 8 output channels this half holds

  __shared__ __align__(16) f16 xs[XN * S2];       // 34560 B  x tile [pos][ch] f16
  __shared__ __align__(16) f16 hs[HN * S2];       // 25920 B  h tile [pos][ch] f16
  __shared__ __align__(16) f16 wsl[9 * 64 * S2];  // 82944 B  weights [tap][o][i] f16
  __shared__ float b1s[64], b2s[64];

  // stage x tile (halo 2, zero-pad at image border), f32 -> f16
  const float* xb = x + (size_t)b * DD;
  for (int i = tid; i < XN * 64; i += 256) {
    int pos = i % XN;
    int ch  = i / XN;
    int rr = pos / XC, cc = pos % XC;
    int gh = oh0 - 2 + rr, gw = ow0 - 2 + cc;
    float v = 0.f;
    if (gh >= 0 && gh < HH && gw >= 0 && gw < WW_)
      v = xb[(size_t)ch * HWW + gh * WW_ + gw];
    xs[pos * S2 + ch] = (f16)v;
  }

  // selected experts + gates (wave-uniform -> scalar loads)
  int go = (br*BB + b) * 2;
  float g_s[2]; int e_s[2];
  g_s[0] = tg[go]; g_s[1] = tg[go+1];
  e_s[0] = ti[go]; e_s[1] = ti[go+1];

  v8f zero = {};
  v8f oacc[4] = {zero, zero, zero, zero};

  for (int s = 0; s < 2; s++) {
    int   eg = br * EE + e_s[s];
    float g  = g_s[s];
    __syncthreads();
    // stage conv1 weights (rows of 64 halfs -> stride-S2 rows) + biases
    const f16* pw1 = pw + (size_t)(eg*2 + 0) * 9 * 64 * 64;
#if HAVE_ASYNC_LDS
    for (int i = tid; i < 4608; i += 256) {
      int row = i >> 3, chunk = i & 7;
      __builtin_amdgcn_global_load_async_to_lds_b128(
          (g_i4v*)(pw1 + (size_t)i * 8),
          (l_i4v*)&wsl[row * S2 + chunk * 8], 0, 0);
    }
#else
    for (int i = tid; i < 4608; i += 256) {
      int row = i >> 3, chunk = i & 7;
      *(uint4*)&wsl[row * S2 + chunk * 8] = ((const uint4*)pw1)[i];
    }
#endif
    if (tid < 64)        b1s[tid]      = be1[eg*64 + tid];
    else if (tid < 128)  b2s[tid - 64] = be2[eg*64 + (tid - 64)];
#if HAVE_ASYNC_LDS
    __builtin_amdgcn_s_wait_asynccnt(0);
#endif
    __syncthreads();

    // conv1: D[ch][hpos] over 4 ch-tiles x 12 pos-tiles = 48 tiles, 6 per wave
    for (int j = 0; j < 6; j++) {
      int t  = wave * 6 + j;
      int mt = t / 12, nt = t % 12;
      int p  = nt * 16 + l;
      int pcl = p < HN ? p : HN - 1;     // clamp (stores are guarded)
      int prr = pcl / HC, pcc = pcl % HC;
      int chA = mt * 16 + l;
      v8f acc = {};
      #pragma unroll
      for (int tap = 0; tap < 9; tap++) {
        int kh = tap / 3, kw = tap % 3;
        const f16* bcol = &xs[((prr + kh) * XC + pcc + kw) * S2];
        const f16* arow = &wsl[(tap * 64 + chA) * S2];
        #pragma unroll
        for (int kc = 0; kc < 64; kc += 32) {
          v16h a  = load_fragA(arow, kc, lane);
          v16h bm = load_fragB(bcol, kc, lane);
          acc = __builtin_amdgcn_wmma_f32_16x16x32_f16(false, a, false, bm,
                                                       (short)0, acc, false, false);
        }
      }
      if (p < HN) {
        int ch0 = mt * 16 + hi8;
        union { f16 e[8]; uint4 q; } pk;
        #pragma unroll
        for (int r = 0; r < 8; r++) {
          float v = acc[r] + b1s[ch0 + r];
          v = v > 0.f ? v : 0.2f * v;          // leaky_relu 0.2
          pk.e[r] = (f16)v;
        }
        *(uint4*)&hs[p * S2 + ch0] = pk.q;     // one ds_store_b128
      }
    }
    __syncthreads();
    // stage conv2 weights (reuse wsl)
    const f16* pw2 = pw + (size_t)(eg*2 + 1) * 9 * 64 * 64;
#if HAVE_ASYNC_LDS
    for (int i = tid; i < 4608; i += 256) {
      int row = i >> 3, chunk = i & 7;
      __builtin_amdgcn_global_load_async_to_lds_b128(
          (g_i4v*)(pw2 + (size_t)i * 8),
          (l_i4v*)&wsl[row * S2 + chunk * 8], 0, 0);
    }
    __builtin_amdgcn_s_wait_asynccnt(0);
#else
    for (int i = tid; i < 4608; i += 256) {
      int row = i >> 3, chunk = i & 7;
      *(uint4*)&wsl[row * S2 + chunk * 8] = ((const uint4*)pw2)[i];
    }
#endif
    __syncthreads();

    // conv2: D[ch][outpos] over 4 ch-tiles x 8 row-tiles = 32 tiles, 4 per wave
    for (int j = 0; j < 4; j++) {
      int t  = wave * 4 + j;
      int mt = t >> 3, nt = t & 7;       // nt == output row of the 8x16 patch
      int chA = mt * 16 + l;
      v8f acc = {};
      #pragma unroll
      for (int tap = 0; tap < 9; tap++) {
        int kh = tap / 3, kw = tap % 3;
        const f16* bcol = &hs[((nt + kh) * HC + l + kw) * S2];
        const f16* arow = &wsl[(tap * 64 + chA) * S2];
        #pragma unroll
        for (int kc = 0; kc < 64; kc += 32) {
          v16h a  = load_fragA(arow, kc, lane);
          v16h bm = load_fragB(bcol, kc, lane);
          acc = __builtin_amdgcn_wmma_f32_16x16x32_f16(false, a, false, bm,
                                                       (short)0, acc, false, false);
        }
      }
      int ch0 = mt * 16 + hi8;
      #pragma unroll
      for (int r = 0; r < 8; r++)
        oacc[j][r] += g * (acc[r] + b2s[ch0 + r]);   // gate-weighted combine
    }
  }

  // store branch output as f16, layout [br][b][h][w][c] (channel-contiguous)
  for (int j = 0; j < 4; j++) {
    int t  = wave * 4 + j;
    int mt = t >> 3, nt = t & 7;
    int ch0 = mt * 16 + hi8;
    int gh = oh0 + nt, gw = ow0 + l;
    union { f16 e[8]; uint4 q; } pk;
    #pragma unroll
    for (int r = 0; r < 8; r++) pk.e[r] = (f16)oacc[j][r];
    *(uint4*)&mid[((size_t)((br*BB + b) * HWW + gh * WW_ + gw)) * 64 + ch0] = pk.q;
  }
}

// ---------------- Kernel 4: reduce conv (concat 128ch -> 64ch, 3x3) ----------
#define X3S 136   // padded LDS row stride (halfs): 272B -> conflict-free frags
__global__ __launch_bounds__(256) void moe_fuse_kernel(
    const f16* __restrict__ mid, const f16* __restrict__ pwr,
    const float* __restrict__ brb, float* __restrict__ out)
{
  int blk = blockIdx.x;
  int b   = blk >> 5;
  int pi  = blk & 31;
  int oh0 = (pi >> 2) * PH;
  int ow0 = (pi & 3) * PW;
  int tid = threadIdx.x, wave = tid >> 5, lane = tid & 31;
  int l = lane & 15, hi8 = (lane >> 4) << 3;

  __shared__ __align__(16) f16 x3[HN * X3S];        // 48960 B [pos][ch128]
  __shared__ __align__(16) f16 w3[3 * 64 * X3S];    // 52224 B 3 taps [o][i128]
  __shared__ float brs[64];

  // stage input tile: concat(rgb, ir) channels, halo 1, zero-pad border
  for (int i = tid; i < HN * 16; i += 256) {
    int pos = i >> 4, chunk = i & 15;
    int rr = pos / HC, cc = pos % HC;
    int gh = oh0 - 1 + rr, gw = ow0 - 1 + cc;
    uint4 v = {0u, 0u, 0u, 0u};
    if (gh >= 0 && gh < HH && gw >= 0 && gw < WW_) {
      int brn = chunk >> 3;               // 0: rgb half, 1: ir half
      int ch  = (chunk & 7) * 8;
      v = *(const uint4*)&mid[((size_t)((brn*BB + b) * HWW + gh * WW_ + gw)) * 64 + ch];
    }
    *(uint4*)&x3[pos * X3S + chunk * 8] = v;
  }
  if (tid < 64) brs[tid] = brb[tid];

  v8f zero = {};
  v8f oacc[4] = {zero, zero, zero, zero};

  for (int g = 0; g < 3; g++) {           // K staged in 3 tap-groups
    __syncthreads();
#if HAVE_ASYNC_LDS
    for (int i = tid; i < 3 * 64 * 16; i += 256) {
      int row = i >> 4, chunk = i & 15;   // row = tl*64 + o
      __builtin_amdgcn_global_load_async_to_lds_b128(
          (g_i4v*)(pwr + (size_t)(192 * g + row) * 128 + chunk * 8),
          (l_i4v*)&w3[row * X3S + chunk * 8], 0, 0);
    }
    __builtin_amdgcn_s_wait_asynccnt(0);
#else
    for (int i = tid; i < 3 * 64 * 16; i += 256) {
      int row = i >> 4, chunk = i & 15;   // row = tl*64 + o
      ((uint4*)&w3[row * X3S])[chunk] =
          ((const uint4*)&pwr[(size_t)(192 * g + row) * 128])[chunk];
    }
#endif
    __syncthreads();
    for (int j = 0; j < 4; j++) {
      int t  = wave * 4 + j;
      int mt = t >> 3, nt = t & 7;
      int chA = mt * 16 + l;
      v8f acc = oacc[j];
      #pragma unroll
      for (int tl = 0; tl < 3; tl++) {
        int tap = 3 * g + tl;
        int kh = tap / 3, kw = tap % 3;
        const f16* bcol = &x3[((nt + kh) * HC + l + kw) * X3S];
        const f16* arow = &w3[(tl * 64 + chA) * X3S];
        #pragma unroll
        for (int kc = 0; kc < 128; kc += 32) {
          v16h a  = load_fragA(arow, kc, lane);
          v16h bm = load_fragB(bcol, kc, lane);
          acc = __builtin_amdgcn_wmma_f32_16x16x32_f16(false, a, false, bm,
                                                       (short)0, acc, false, false);
        }
      }
      oacc[j] = acc;
    }
  }

  // final output: f32 NCHW + bias
  for (int j = 0; j < 4; j++) {
    int t  = wave * 4 + j;
    int mt = t >> 3, nt = t & 7;
    int ch0 = mt * 16 + hi8;
    int gh = oh0 + nt, gw = ow0 + l;
    #pragma unroll
    for (int r = 0; r < 8; r++)
      out[(size_t)(b * CC + ch0 + r) * HWW + gh * WW_ + gw] = oacc[j][r] + brs[ch0 + r];
  }
}

// ---------------- launch -----------------------------------------------------
extern "C" void kernel_launch(void* const* d_in, const int* in_sizes, int n_in,
                              void* d_out, int out_size, void* d_ws, size_t ws_size,
                              hipStream_t stream)
{
  const float* xr  = (const float*)d_in[0];
  const float* xi  = (const float*)d_in[1];
  const float* nr  = (const float*)d_in[2];
  const float* ni  = (const float*)d_in[3];
  const float* wgr = (const float*)d_in[4];
  const float* wnr = (const float*)d_in[5];
  const float* wgi = (const float*)d_in[6];
  const float* wni = (const float*)d_in[7];
  const float* We1 = (const float*)d_in[8];
  const float* be1 = (const float*)d_in[9];
  const float* We2 = (const float*)d_in[10];
  const float* be2 = (const float*)d_in[11];
  const float* Wr  = (const float*)d_in[12];
  const float* brb = (const float*)d_in[13];

  char* ws  = (char*)d_ws;
  float* tg = (float*)(ws + WS_TG);
  int*   ti = (int*)(ws + WS_TI);
  f16*   pw = (f16*)(ws + WS_PW);
  f16*   pwr= (f16*)(ws + WS_PWR);
  f16*   mid= (f16*)(ws + WS_MID);

  moe_gate_kernel<<<64, 256, 0, stream>>>(xr, xi, nr, ni, wgr, wnr, wgi, wni, tg, ti);
  moe_pack_kernel<<<(PW_HALFS + PWR_HALFS) / 256, 256, 0, stream>>>(We1, We2, Wr, pw, pwr);
  moe_expert_kernel<<<2048, 256, 0, stream>>>(xr, xi, be1, be2, pw, tg, ti, mid);
  moe_fuse_kernel<<<1024, 256, 0, stream>>>(mid, pwr, brb, (float*)d_out);
}
